// CTCDecoder_31576599561102
// MI455X (gfx1250) — compile-verified, matrix operand order checked by reading, not run on hardware
//
#include <hip/hip_runtime.h>
#include <hip/hip_bf16.h>

// ---------- types ----------
typedef __bf16 bf16_t;
typedef bf16_t v8bf  __attribute__((ext_vector_type(8)));
typedef bf16_t v16bf __attribute__((ext_vector_type(16)));
typedef float  v8f   __attribute__((ext_vector_type(8)));
typedef float  v4f   __attribute__((ext_vector_type(4)));

// Problem constants (from reference): B=64, H=8, W=512, C=64 -> D=512, U=256
#define BATCH 64
#define TSEQ  512
#define DIN   512
#define UHID  256
#define NGATE 1024              // 4*U
#define KTOT  768               // D + U
#define MROWS (TSEQ * BATCH)    // 32768

// Workspace layout (bytes)
#define OFF_FEATS 0ull                       // bf16 [MROWS][DIN]                = 32 MB
#define OFF_WT    33554432ull                // bf16 [2][NGATE][KTOT]            = 3 MB
#define OFF_ZX    36700160ull                // f32  [2][TSEQ][BATCH][NGATE]     = 256 MB
#define OFF_HBUF  305135616ull               // bf16 [2 dir][2 ping][64][256]    = 128 KB
#define OFF_BAR   305266688ull               // u32  [2 dir][2] {count, phase}

__device__ __forceinline__ v16bf concat8(v8bf lo, v8bf hi) {
  return __builtin_shufflevector(lo, hi, 0,1,2,3,4,5,6,7,8,9,10,11,12,13,14,15);
}
__device__ __forceinline__ v8f wmma_bf16(v16bf a, v16bf b, v8f c) {
  // D = A(16x32 bf16) x B(32x16 bf16) + C(16x16 f32)
  return __builtin_amdgcn_wmma_f32_16x16x32_bf16(false, a, false, b, (short)0, c, false, false);
}
__device__ __forceinline__ float sigmoidf_fast(float x) {
  return 1.0f / (1.0f + __expf(-x));
}

// ---------- kernel 0: zero the cross-workgroup barrier state ----------
__global__ void init_bar_kernel(unsigned int* __restrict__ bar) {
  if (threadIdx.x < 4) bar[threadIdx.x] = 0u;
}

// ---------- kernel 1: W [768,1024] f32 -> Wt [1024,768] bf16 (N-major) ----------
__global__ void prep_weights_kernel(const float* __restrict__ Wfw,
                                    const float* __restrict__ Wbw,
                                    bf16_t* __restrict__ Wt) {
  int idx = blockIdx.x * blockDim.x + threadIdx.x;        // over 2*1024*768
  if (idx >= 2 * NGATE * KTOT) return;
  int dir = idx / (NGATE * KTOT);
  int r   = idx % (NGATE * KTOT);
  int n   = r / KTOT;
  int k   = r % KTOT;
  const float* Wsrc = dir ? Wbw : Wfw;
  Wt[idx] = (bf16_t)Wsrc[(size_t)k * NGATE + n];
}

// ---------- kernel 2: x [B,H,T,C] f32 -> feats [(t*64+b), 512] bf16 ----------
__global__ void prep_feats_kernel(const float* __restrict__ x,
                                  bf16_t* __restrict__ feats) {
  int vec = blockIdx.x * blockDim.x + threadIdx.x;        // each handles 4 elems
  size_t e = (size_t)vec * 4;
  if (e >= (size_t)MROWS * DIN) return;
  int row = (int)(e / DIN);            // = t*64 + b
  int col = (int)(e % DIN);            // = h*64 + c
  int t = row / BATCH, b = row % BATCH;
  int h = col / 64,    c = col % 64;
  const float* src = x + (((size_t)b * 8 + h) * TSEQ + t) * 64 + c;  // contiguous in c
  v4f v = *(const v4f*)src;
  bf16_t* dst = feats + e;
  dst[0] = (bf16_t)v[0]; dst[1] = (bf16_t)v[1];
  dst[2] = (bf16_t)v[2]; dst[3] = (bf16_t)v[3];
}

// ---------- kernel 3: Zx[dir] = feats @ Wx[dir] + bias[dir] ----------
// grid: (MROWS/64, NGATE/128, 2), block: 256 threads = 8 waves (4M x 2N),
// wave tile = 16M x 64N (4 WMMA subtiles), K = DIN = 512.
__global__ __launch_bounds__(256) void gemm_zx_kernel(const bf16_t* __restrict__ feats,
                                                      const bf16_t* __restrict__ Wt,
                                                      const float* __restrict__ bfw,
                                                      const float* __restrict__ bbw,
                                                      float* __restrict__ Zx) {
  __shared__ alignas(16) bf16_t As[64 * 32];    // 4 KB
  __shared__ alignas(16) bf16_t Bs[128 * 32];   // 8 KB

  const int dir = blockIdx.z;
  const int m0  = blockIdx.x * 64;
  const int n0  = blockIdx.y * 128;
  const bf16_t* wt = Wt + (size_t)dir * NGATE * KTOT;       // [1024][768], x-part at k<512
  const float*  bias = dir ? bbw : bfw;
  float* zxd = Zx + (size_t)dir * MROWS * NGATE;

  const int tid   = threadIdx.x;
  const int wave  = tid >> 5;
  const int lane  = tid & 31;
  const int l16   = lane & 15;
  const int lhalf = lane >> 4;
  const int wm    = wave >> 1;   // 0..3  (M subtile)
  const int wn    = wave & 1;    // 0..1  (N half: 64 cols)

  v8f acc[4];
  for (int s = 0; s < 4; ++s) acc[s] = (v8f)0.0f;

  for (int kc = 0; kc < DIN / 32; ++kc) {
    const int k0 = kc * 32;
    {
      int r = tid >> 2, c4 = tid & 3;
      *(v8bf*)&As[r * 32 + c4 * 8] =
          *(const v8bf*)&feats[(size_t)(m0 + r) * DIN + k0 + c4 * 8];
    }
    {
      int r = tid >> 1, h = tid & 1;
      const bf16_t* src = wt + (size_t)(n0 + r) * KTOT + k0 + h * 16;
      *(v8bf*)&Bs[r * 32 + h * 16]     = *(const v8bf*)src;
      *(v8bf*)&Bs[r * 32 + h * 16 + 8] = *(const v8bf*)(src + 8);
    }
    __syncthreads();

    const bf16_t* ap = &As[(wm * 16 + l16) * 32 + lhalf * 8];
    v16bf afrag = concat8(*(const v8bf*)ap, *(const v8bf*)(ap + 16));
#pragma unroll
    for (int s = 0; s < 4; ++s) {
      const bf16_t* bp = &Bs[(wn * 64 + s * 16 + l16) * 32 + lhalf * 8];
      v16bf bfrag = concat8(*(const v8bf*)bp, *(const v8bf*)(bp + 16));
      acc[s] = wmma_bf16(afrag, bfrag, acc[s]);
    }
    __syncthreads();
  }

#pragma unroll
  for (int s = 0; s < 4; ++s) {
    int nglob = n0 + wn * 64 + s * 16 + l16;
    float bv = bias[nglob];
#pragma unroll
    for (int r = 0; r < 8; ++r) {
      int mglob = m0 + wm * 16 + lhalf * 8 + r;
      zxd[(size_t)mglob * NGATE + nglob] = acc[s][r] + bv;
    }
  }
}

// ---------- kernel 4: recurrent scan, 4 workgroups per direction ----------
// Block bx: dir = bx>>2, chunk = bx&3 owns u in [chunk*64, chunk*64+64).
// 512 threads = 16 waves; wave w: batch tile mt=w&3 (16 rows), u-tile uu=w>>2.
// Each wave computes all 4 gate tiles for its (mt,uu) -> lane-local cell update.
// W_h slice for this chunk (256 N cols x 256 K, bf16 = 128 KB) is pinned in LDS
// for the whole scan; full h (bf16, 32 KB) mirrored in LDS. Cross-workgroup
// h exchange via ping-pong global buffers + device-scope phase barrier.
__global__ __launch_bounds__(512) void lstm_scan_kernel(const float* __restrict__ Zx,
                                                        const bf16_t* __restrict__ Wt,
                                                        bf16_t* __restrict__ hbuf,
                                                        unsigned int* __restrict__ bar,
                                                        float* __restrict__ out) {
  __shared__ alignas(16) bf16_t h_lds[BATCH * UHID];        // 32 KB (full h)
  __shared__ alignas(16) bf16_t w_lds[4 * 64 * UHID];       // 128 KB (W_h slice)

  const int dir   = blockIdx.x >> 2;
  const int chunk = blockIdx.x & 3;
  const float*  zx = Zx + (size_t)dir * MROWS * NGATE;            // [T][64][1024]
  const bf16_t* wt = Wt + (size_t)dir * NGATE * KTOT;             // [1024][768]
  unsigned int* bcount = &bar[dir * 2 + 0];
  unsigned int* bphase = &bar[dir * 2 + 1];

  const int tid   = threadIdx.x;
  const int wave  = tid >> 5;
  const int lane  = tid & 31;
  const int l16   = lane & 15;
  const int lhalf = lane >> 4;
  const int mt    = wave & 3;    // batch tile (16 rows)
  const int uu    = wave >> 2;   // 0..3, u-tile within chunk

  // ---- load W_h slice into LDS: row (g*64 + ulocal), col k (0..255) ----
  for (int j = tid; j < (4 * 64 * UHID) / 8; j += 512) {     // v8bf units
    int row = j >> 5;            // 0..255
    int kv  = j & 31;
    int g   = row >> 6;
    int ul  = row & 63;
    int ng  = g * UHID + chunk * 64 + ul;                    // global N column
    *(v8bf*)&w_lds[row * UHID + kv * 8] =
        *(const v8bf*)&wt[(size_t)ng * KTOT + DIN + kv * 8]; // h-part: k>=512
  }
  for (int i = tid; i < BATCH * UHID; i += 512) h_lds[i] = (bf16_t)0.0f;
  __syncthreads();

  float cstate[8];
#pragma unroll
  for (int i = 0; i < 8; ++i) cstate[i] = 0.0f;

  const int u_local = uu * 16 + l16;
  const int u_glob  = chunk * 64 + u_local;

  for (int step = 0; step < TSEQ; ++step) {
    const int t = dir ? (TSEQ - 1 - step) : step;
    const float* zstep = zx + (size_t)t * BATCH * NGATE;

    // prefetch next step's Zx slab (256 KB) while we compute this one
    if (step + 1 < TSEQ) {
      const int tn = dir ? (t - 1) : (t + 1);
      __builtin_prefetch(zx + (size_t)tn * BATCH * NGATE + tid * 128, 0, 1);
    }

    // init accumulators from precomputed Zx (bias already folded in)
    v8f acc[4];
#pragma unroll
    for (int g = 0; g < 4; ++g) {
      int n = g * UHID + u_glob;
#pragma unroll
      for (int r = 0; r < 8; ++r)
        acc[g][r] = zstep[(size_t)(mt * 16 + lhalf * 8 + r) * NGATE + n];
    }

    // z += h @ W_h   (K = 256, 8 chunks of 32) -- pure LDS inner loop
#pragma unroll
    for (int kc = 0; kc < UHID / 32; ++kc) {
      const int k0 = kc * 32;
      const bf16_t* ap = &h_lds[(mt * 16 + l16) * UHID + k0 + lhalf * 8];
      v16bf afrag = concat8(*(const v8bf*)ap, *(const v8bf*)(ap + 16));
#pragma unroll
      for (int g = 0; g < 4; ++g) {
        const bf16_t* bp = &w_lds[(g * 64 + u_local) * UHID + k0 + lhalf * 8];
        v16bf bfrag = concat8(*(const v8bf*)bp, *(const v8bf*)(bp + 16));
        acc[g] = wmma_bf16(afrag, bfrag, acc[g]);
      }
    }

    // gates + state update (TF order i, j, f, o; forget bias = 1.0)
    bf16_t* hb = hbuf + ((size_t)dir * 2 + (step & 1)) * (BATCH * UHID);
#pragma unroll
    for (int r = 0; r < 8; ++r) {
      int b = mt * 16 + lhalf * 8 + r;
      float c_old = cstate[r];
      float cn = sigmoidf_fast(acc[2][r] + 1.0f) * c_old +
                 sigmoidf_fast(acc[0][r]) * tanhf(acc[1][r]);
      float hn = sigmoidf_fast(acc[3][r]) * tanhf(cn);
      cstate[r] = cn;
      hb[b * UHID + u_glob] = (bf16_t)hn;
      out[(size_t)b * (TSEQ * 2 * UHID) + (size_t)t * (2 * UHID) + dir * UHID + u_glob] = hn;
    }

    // ---- device-scope barrier across the 4 blocks of this direction ----
    __threadfence();            // make this block's h-chunk globally visible
    __syncthreads();
    if (tid == 0) {
      unsigned int old = __hip_atomic_fetch_add(bcount, 1u, __ATOMIC_ACQ_REL,
                                                __HIP_MEMORY_SCOPE_AGENT);
      if (old == 3u) {
        __hip_atomic_store(bcount, 0u, __ATOMIC_RELAXED, __HIP_MEMORY_SCOPE_AGENT);
        __hip_atomic_fetch_add(bphase, 1u, __ATOMIC_RELEASE, __HIP_MEMORY_SCOPE_AGENT);
      } else {
        while (__hip_atomic_load(bphase, __ATOMIC_ACQUIRE, __HIP_MEMORY_SCOPE_AGENT)
               < (unsigned int)(step + 1)) {
          __builtin_amdgcn_s_sleep(1);
        }
      }
    }
    __syncthreads();
    __threadfence();            // discard stale ping-buffer lines before reading

    // reload full h (all 4 chunks) from the ping buffer into LDS
    for (int j = tid; j < (BATCH * UHID) / 8; j += 512) {
      *(v8bf*)&h_lds[j * 8] = *(const v8bf*)&hb[j * 8];
    }
    __syncthreads();
  }
}

// ---------- launcher ----------
extern "C" void kernel_launch(void* const* d_in, const int* in_sizes, int n_in,
                              void* d_out, int out_size, void* d_ws, size_t ws_size,
                              hipStream_t stream) {
  (void)in_sizes; (void)n_in; (void)out_size; (void)ws_size;
  const float* x   = (const float*)d_in[0];   // [64,8,512,64]
  const float* Wfw = (const float*)d_in[1];   // [768,1024]
  const float* bfw = (const float*)d_in[2];   // [1024]
  const float* Wbw = (const float*)d_in[3];   // [768,1024]
  const float* bbw = (const float*)d_in[4];   // [1024]
  float* out = (float*)d_out;                 // [64,512,512]

  char* ws = (char*)d_ws;
  bf16_t* feats = (bf16_t*)(ws + OFF_FEATS);
  bf16_t* Wt    = (bf16_t*)(ws + OFF_WT);
  float*  Zx    = (float*) (ws + OFF_ZX);
  bf16_t* hbuf  = (bf16_t*)(ws + OFF_HBUF);
  unsigned int* bar = (unsigned int*)(ws + OFF_BAR);

  // 0) reset cross-workgroup barrier (deterministic across replays)
  init_bar_kernel<<<1, 64, 0, stream>>>(bar);
  // 1) weights -> bf16, transposed to N-major
  {
    int total = 2 * NGATE * KTOT;
    prep_weights_kernel<<<(total + 255) / 256, 256, 0, stream>>>(Wfw, Wbw, Wt);
  }
  // 2) x -> bf16 feats [(t*64+b), 512]
  {
    int total_vec4 = (MROWS * DIN) / 4;
    prep_feats_kernel<<<(total_vec4 + 255) / 256, 256, 0, stream>>>(x, feats);
  }
  // 3) Zx = feats @ Wx + b for both directions
  {
    dim3 grid(MROWS / 64, NGATE / 128, 2);
    gemm_zx_kernel<<<grid, 256, 0, stream>>>(feats, Wt, bfw, bbw, Zx);
  }
  // 4) recurrent scan: 4 cooperating workgroups per direction
  lstm_scan_kernel<<<8, 512, 0, stream>>>(Zx, Wt, hbuf, bar, out);
}
